// Head_43293270344024
// MI455X (gfx1250) — matrix operand order repaired and checked by reading, not compiled, not run on hardware
//
#include <hip/hip_runtime.h>

#define CTX   2048
#define NEMBD 1024
#define HEAD  128
#define BATCH 4

typedef __attribute__((ext_vector_type(16))) __bf16 v16bf;
typedef __attribute__((ext_vector_type(8)))  float  v8f;
typedef __attribute__((ext_vector_type(4)))  unsigned int v4u;
typedef __attribute__((ext_vector_type(4)))  int    v4i;
typedef __attribute__((ext_vector_type(8)))  int    v8i;
typedef unsigned short bfu;
typedef unsigned int   u32t;

#if defined(__has_builtin)
#if __has_builtin(__builtin_amdgcn_tensor_load_to_lds) && __has_builtin(__builtin_amdgcn_s_wait_tensorcnt)
#define USE_TDM 1
#if __has_include(<hip/amd_detail/amd_gfx1250_TDM.h>)
#define TDM_6ARG 1
#endif
#endif
#endif

__device__ __forceinline__ bfu f2bf(float f) {
  u32t u = __builtin_bit_cast(u32t, f);
  u32t r = (u + 0x7FFFu + ((u >> 16) & 1u)) >> 16;   // round-to-nearest-even
  return (bfu)r;
}

union FragB16 { v16bf v; bfu s[16]; uint4 q[2]; };

__device__ __forceinline__ v8f vzero8() {
  v8f v;
#pragma unroll
  for (int i = 0; i < 8; ++i) v[i] = 0.0f;
  return v;
}

// B-fragment lane load: 16 contiguous bf16 (32B, two b128 loads).
__device__ __forceinline__ v16bf ldB(const bfu* p) {
  FragB16 u;
  u.q[0] = *(const uint4*)(p);
  u.q[1] = *(const uint4*)(p + 8);
  return u.v;
}

// A-fragment (16x32 bf16, MxK): K = kh*16 + (0..15), two 16B chunks.
__device__ __forceinline__ v16bf ldA(const bfu* row, int c0, int kh) {
  FragB16 u;
  u.q[0] = *(const uint4*)(row + c0 + kh * 16);
  u.q[1] = *(const uint4*)(row + c0 + kh * 16 + 8);
  return u.v;
}

// A-fragment from fp32 row (convert to bf16 in-flight).
__device__ __forceinline__ v16bf ldAf32(const float* row, int c0, int kh) {
  FragB16 u;
#pragma unroll
  for (int h = 0; h < 2; ++h) {
    const float* p = row + c0 + kh * 16 + h * 8;
    float4 f0 = *(const float4*)(p);
    float4 f1 = *(const float4*)(p + 4);
    int b = h * 8;
    u.s[b + 0] = f2bf(f0.x); u.s[b + 1] = f2bf(f0.y);
    u.s[b + 2] = f2bf(f0.z); u.s[b + 3] = f2bf(f0.w);
    u.s[b + 4] = f2bf(f1.x); u.s[b + 5] = f2bf(f1.y);
    u.s[b + 6] = f2bf(f1.z); u.s[b + 7] = f2bf(f1.w);
  }
  return u.v;
}

#define WMMA_BF16(A, B, C) \
  __builtin_amdgcn_wmma_f32_16x16x32_bf16(false, (A), false, (B), (short)0, (C), false, false)

#ifdef USE_TDM
// 2D TDM load: tile [tile_d1 rows x tile_d0 elems] of a row-major bf16 tensor
// [tensor_d1 x tensor_d0] (row stride stride0 elems) -> contiguous LDS tile.
// Descriptor per CDNA5 ISA §8 (data_size=1 -> 2 bytes).
__device__ __forceinline__ void tdm_load_2d(unsigned lds_addr, const void* gaddr,
                                            unsigned tensor_d0, unsigned tensor_d1,
                                            unsigned tile_d0, unsigned tile_d1,
                                            unsigned stride0) {
  unsigned long long ga = (unsigned long long)(size_t)gaddr;
  v4u g0;
  g0[0] = 1u;                                           // count=1, user mode
  g0[1] = lds_addr;                                     // lds_addr [63:32]
  g0[2] = (unsigned)ga;                                 // global_addr low
  g0[3] = (unsigned)((ga >> 32) & 0x01FFFFFFu) | (2u << 30);  // addr hi | type=2
  v8i g1;
  g1[0] = (int)(1u << 16);                              // data_size=1 (2B)
  g1[1] = (int)(tensor_d0 << 16);                       // tensor_dim0 lo16
  g1[2] = (int)((tensor_d0 >> 16) | (tensor_d1 << 16)); // dim0 hi | dim1 lo
  g1[3] = (int)((tensor_d1 >> 16) | (tile_d0 << 16));   // dim1 hi | tile_dim0
  g1[4] = (int)(tile_d1 & 0xFFFFu);                     // tile_dim1, tile_dim2=0
  g1[5] = (int)stride0;                                 // dim0_stride lo32
  g1[6] = 0;
  g1[7] = 0;
  v4i z4 = {0, 0, 0, 0};
#ifdef TDM_6ARG
  v8i z8 = {0, 0, 0, 0, 0, 0, 0, 0};
  __builtin_amdgcn_tensor_load_to_lds(g0, g1, z4, z4, z8, 0);
#else
  __builtin_amdgcn_tensor_load_to_lds(g0, g1, z4, z4, 0);
#endif
}
#endif

// ---------------------------------------------------------------------------
// Kernel 0: convert+transpose weights fp32 [1024][128] -> bf16 [3][128][1024]
// ---------------------------------------------------------------------------
__global__ void k0_wt(const float* __restrict__ Wq, const float* __restrict__ Wk,
                      const float* __restrict__ Wv, bfu* __restrict__ wt) {
  int tid = blockIdx.x * 256 + threadIdx.x;   // 3*128*1024 total
  int m   = tid >> 17;
  int rem = tid & ((1 << 17) - 1);
  int n   = rem >> 10;
  int c   = rem & 1023;
  const float* W = (m == 0) ? Wq : (m == 1) ? Wk : Wv;
  wt[tid] = f2bf(W[c * HEAD + n]);            // wt[m][n][c] = W[c][n]
}

// ---------------------------------------------------------------------------
// Kernel 1: q/k/v = x @ W via WMMA, software-pipelined over the K dimension.
// Outputs: q,k row-major bf16 [B*T][128]; v transposed bf16 [B][128][T].
// ---------------------------------------------------------------------------
__global__ __launch_bounds__(128) void k1_qkv(const float* __restrict__ x,
                       const bfu* __restrict__ wt,
                       bfu* __restrict__ qg, bfu* __restrict__ kg,
                       bfu* __restrict__ vtg) {
  const int lane = threadIdx.x & 31;
  const int wv   = threadIdx.x >> 5;
  const int lm   = lane & 15;
  const int kh   = lane >> 4;
  const int rowTile = blockIdx.x * 4 + wv;    // 0..511 (B*T/16)
  const int mat  = blockIdx.y;                // 0=q 1=k 2=v
  const bfu* W = wt + (size_t)mat * HEAD * NEMBD;            // [128][1024]
  const float* xr = x + (size_t)(rowTile * 16 + lm) * NEMBD;

  v8f acc[8];
#pragma unroll
  for (int h = 0; h < 8; ++h) acc[h] = vzero8();

  v16bf a_cur = ldAf32(xr, 0, kh);
  v16bf b_cur[8];
#pragma unroll
  for (int h = 0; h < 8; ++h)
    b_cur[h] = ldB(W + (size_t)(h * 16 + lm) * NEMBD + kh * 16);

  for (int it = 0; it < (NEMBD / 32) - 1; ++it) {
    const int c1 = (it + 1) * 32;
    v16bf a_nxt = ldAf32(xr, c1, kh);
    v16bf b_nxt[8];
#pragma unroll
    for (int h = 0; h < 8; ++h)
      b_nxt[h] = ldB(W + (size_t)(h * 16 + lm) * NEMBD + c1 + kh * 16);
#pragma unroll
    for (int h = 0; h < 8; ++h)
      acc[h] = WMMA_BF16(a_cur, b_cur[h], acc[h]);
    a_cur = a_nxt;
#pragma unroll
    for (int h = 0; h < 8; ++h) b_cur[h] = b_nxt[h];
  }
#pragma unroll
  for (int h = 0; h < 8; ++h)
    acc[h] = WMMA_BF16(a_cur, b_cur[h], acc[h]);

#pragma unroll
  for (int h = 0; h < 8; ++h) {
#pragma unroll
    for (int r = 0; r < 8; ++r) {
      int row = rowTile * 16 + r + 8 * kh;     // C-frag: row = r + 8*kh
      int col = h * 16 + lm;
      bfu bits = f2bf(acc[h][r]);
      if (mat == 0)      qg[(size_t)row * HEAD + col] = bits;
      else if (mat == 1) kg[(size_t)row * HEAD + col] = bits;
      else {
        int bb = row >> 11, t = row & (CTX - 1);
        vtg[(size_t)bb * HEAD * CTX + (size_t)col * CTX + t] = bits;
      }
    }
  }
}

// ---------------------------------------------------------------------------
// Kernel 2: fused causal flash attention.
// 256 threads = 8 waves; wave w owns Q rows [q0+16w, q0+16w+16).
// K/V tiles double-buffered in LDS; the Tensor Data Mover prefetches tile
// j+1 while tile j is being consumed (s_wait_tensorcnt 2 keeps it in flight).
// ---------------------------------------------------------------------------
__global__ __launch_bounds__(256) void k2_attn(const bfu* __restrict__ qg,
                        const bfu* __restrict__ kg,
                        const bfu* __restrict__ vtg,
                        float* __restrict__ out) {
  __shared__ __align__(16) bfu lds_k [2][32 * HEAD];   // [buf][key][c]
  __shared__ __align__(16) bfu lds_vt[2][HEAD * 32];   // [buf][c][key]
  __shared__ __align__(16) bfu lds_p [8][16 * 32];     // per-wave P [m][key]

  const int lane = threadIdx.x & 31;
  const int wv   = threadIdx.x >> 5;
  const int lm   = lane & 15;
  const int kh   = lane >> 4;
  const int q0   = blockIdx.x * 128;
  const int b    = blockIdx.y;

  const bfu* qb  = qg  + (size_t)b * CTX * HEAD;
  const bfu* kb  = kg  + (size_t)b * CTX * HEAD;
  const bfu* vtb = vtg + (size_t)b * HEAD * CTX;

  const int   qrow  = q0 + wv * 16;
  const float scale = 0.08838834764831845f;   // 1/sqrt(128)

  v16bf qf[4];
#pragma unroll
  for (int cc = 0; cc < 4; ++cc)
    qf[cc] = ldA(qb + (size_t)(qrow + lm) * HEAD, cc * 32, kh);

  v8f acc[8];
#pragma unroll
  for (int h = 0; h < 8; ++h) acc[h] = vzero8();
  float mi[8], li[8];
#pragma unroll
  for (int r = 0; r < 8; ++r) { mi[r] = -__builtin_inff(); li[r] = 0.0f; }

  const int nkb = (q0 + 128) / 32;

#ifdef USE_TDM
  const unsigned ldsKA  = (unsigned)(size_t)&lds_k[0][0];
  const unsigned ldsVtA = (unsigned)(size_t)&lds_vt[0][0];
  const unsigned kBufBytes  = 32 * HEAD * 2;
  const unsigned vtBufBytes = HEAD * 32 * 2;
  if (threadIdx.x < 32) {                      // prologue: tile 0 in flight
    tdm_load_2d(ldsKA,  kb,  HEAD, CTX, HEAD, 32, HEAD);
    tdm_load_2d(ldsVtA, vtb, CTX,  HEAD, 32, HEAD, CTX);
  }
#endif

  for (int j = 0; j < nkb; ++j) {
#ifdef USE_TDM
    if (threadIdx.x < 32) {                    // wave 0 drives the TDM
      if (j + 1 < nkb) {                       // prefetch tile j+1 (other buf)
        const unsigned nb = (unsigned)((j + 1) & 1);
        tdm_load_2d(ldsKA  + nb * kBufBytes,  kb + (size_t)(j + 1) * 32 * HEAD,
                    HEAD, CTX, HEAD, 32, HEAD);
        tdm_load_2d(ldsVtA + nb * vtBufBytes, vtb + (size_t)(j + 1) * 32,
                    CTX, HEAD, 32, HEAD, CTX);
        __builtin_amdgcn_s_wait_tensorcnt(2);  // tile j done; j+1 stays in flight
      } else {
        __builtin_amdgcn_s_wait_tensorcnt(0);  // last tile
      }
    }
    const bfu* kbuf  = &lds_k [j & 1][0];
    const bfu* vtbuf = &lds_vt[j & 1][0];
#else
    const bfu* kbuf  = &lds_k [0][0];
    const bfu* vtbuf = &lds_vt[0][0];
    for (int ch = threadIdx.x; ch < 512; ch += 256) {
      int key = ch >> 4, off = (ch & 15) * 8;
      *(uint4*)&lds_k[0][key * HEAD + off] =
          *(const uint4*)&kb[(size_t)(j * 32 + key) * HEAD + off];
    }
    for (int ch = threadIdx.x; ch < 512; ch += 256) {
      int c = ch >> 2, off = (ch & 3) * 8;
      *(uint4*)&lds_vt[0][c * 32 + off] =
          *(const uint4*)&vtb[(size_t)c * CTX + j * 32 + off];
    }
    if (j + 1 < nkb) {
      __builtin_prefetch(&kb[(size_t)((j + 1) * 32 + (threadIdx.x >> 3)) * HEAD], 0, 0);
      __builtin_prefetch(&vtb[(size_t)(threadIdx.x >> 1) * CTX + (j + 1) * 32], 0, 0);
    }
#endif
    __syncthreads();

    if (j * 32 <= qrow + 15) {        // wave-uniform skip (EXEC stays all-1s)
      // ---- S (16x32) = Q @ K^T : preload 8 B-frags, then 8 WMMAs ----
      v16bf bk[8];
#pragma unroll
      for (int cc = 0; cc < 4; ++cc) {
        bk[cc]     = ldB(kbuf + (size_t)(lm)      * HEAD + cc * 32 + kh * 16);
        bk[4 + cc] = ldB(kbuf + (size_t)(16 + lm) * HEAD + cc * 32 + kh * 16);
      }
      v8f s0 = vzero8(), s1 = vzero8();
#pragma unroll
      for (int cc = 0; cc < 4; ++cc) {
        s0 = WMMA_BF16(qf[cc], bk[cc],     s0);
        s1 = WMMA_BF16(qf[cc], bk[4 + cc], s1);
      }
      // ---- causal mask + online softmax ----
      bfu* pw = &lds_p[wv][0];
#pragma unroll
      for (int r = 0; r < 8; ++r) {
        int row  = qrow + r + 8 * kh;
        int key0 = j * 32 + lm;
        float av = (key0      <= row) ? s0[r] * scale : -__builtin_inff();
        float bv = (key0 + 16 <= row) ? s1[r] * scale : -__builtin_inff();
        float mx = fmaxf(av, bv);
        mx = fmaxf(mx, __shfl_xor(mx, 1, 32));
        mx = fmaxf(mx, __shfl_xor(mx, 2, 32));
        mx = fmaxf(mx, __shfl_xor(mx, 4, 32));
        mx = fmaxf(mx, __shfl_xor(mx, 8, 32));
        float mn = fmaxf(mi[r], mx);
        float al = __expf(mi[r] - mn);
        mi[r] = mn;
        float p0 = __expf(av - mn);
        float p1 = __expf(bv - mn);
        float rs = p0 + p1;
        rs += __shfl_xor(rs, 1, 32);
        rs += __shfl_xor(rs, 2, 32);
        rs += __shfl_xor(rs, 4, 32);
        rs += __shfl_xor(rs, 8, 32);
        li[r] = li[r] * al + rs;
#pragma unroll
        for (int h = 0; h < 8; ++h) acc[h][r] *= al;
        pw[(r + 8 * kh) * 32 + lm]      = f2bf(p0);
        pw[(r + 8 * kh) * 32 + 16 + lm] = f2bf(p1);
      }
      // ---- acc (16x128) += P (16x32) @ V : preload B-frags, 8 WMMAs ----
      v16bf pf = ldA(pw + lm * 32, 0, kh);
      v16bf bv8[8];
#pragma unroll
      for (int h = 0; h < 8; ++h)
        bv8[h] = ldB(vtbuf + (size_t)(h * 16 + lm) * 32 + kh * 16);
#pragma unroll
      for (int h = 0; h < 8; ++h)
        acc[h] = WMMA_BF16(pf, bv8[h], acc[h]);
    }
    __syncthreads();   // all compute on buf j done -> safe to overwrite at j+2
  }

  // ---- normalize + store fp32 ----
#pragma unroll
  for (int r = 0; r < 8; ++r) {
    float inv = 1.0f / li[r];
    int row = qrow + r + 8 * kh;
#pragma unroll
    for (int h = 0; h < 8; ++h)
      out[((size_t)b * CTX + row) * HEAD + h * 16 + lm] = acc[h][r] * inv;
  }
}

// ---------------------------------------------------------------------------
extern "C" void kernel_launch(void* const* d_in, const int* in_sizes, int n_in,
                              void* d_out, int out_size, void* d_ws, size_t ws_size,
                              hipStream_t stream) {
  (void)in_sizes; (void)n_in; (void)out_size; (void)ws_size;
  const float* x  = (const float*)d_in[0];
  const float* Wq = (const float*)d_in[1];
  const float* Wk = (const float*)d_in[2];
  const float* Wv = (const float*)d_in[3];
  float* out = (float*)d_out;

  char* ws = (char*)d_ws;
  const size_t WT_BYTES = (size_t)3 * HEAD * NEMBD * 2;   // 768 KB
  const size_t QK_BYTES = (size_t)BATCH * CTX * HEAD * 2; // 2 MB each
  bfu* wt  = (bfu*)(ws);
  bfu* qg  = (bfu*)(ws + WT_BYTES);
  bfu* kg  = (bfu*)(ws + WT_BYTES + QK_BYTES);
  bfu* vtg = (bfu*)(ws + WT_BYTES + 2 * QK_BYTES);

  k0_wt  <<<dim3((3 * HEAD * NEMBD) / 256), 256, 0, stream>>>(Wq, Wk, Wv, wt);
  k1_qkv <<<dim3(128, 3), 128, 0, stream>>>(x, wt, qg, kg, vtg);
  k2_attn<<<dim3(CTX / 128, BATCH), 256, 0, stream>>>(qg, kg, vtg, out);
}
